// Decoder_55284819034617
// MI455X (gfx1250) — matrix operand order repaired and checked by reading, not compiled, not run on hardware
//
#include <hip/hip_runtime.h>
#include <hip/hip_bf16.h>

typedef __attribute__((ext_vector_type(16))) __bf16 bf16x16;
typedef __attribute__((ext_vector_type(8)))  __bf16 bf16x8;
typedef __attribute__((ext_vector_type(8)))  float  v8f;

#define NEG_INF (-__builtin_inff())

#define CAT16(lo, hi) __builtin_shufflevector(lo, hi, 0,1,2,3,4,5,6,7,8,9,10,11,12,13,14,15)

__device__ __forceinline__ float wave_reduce(float p) {
#pragma unroll
  for (int off = 16; off > 0; off >>= 1) p += __shfl_down(p, off, 32);
  return p;
}

__device__ __forceinline__ float dot4(float4 a, float4 b) {
  return a.x * b.x + a.y * b.y + a.z * b.z + a.w * b.w;
}

// ---------------------------------------------------------------- prep kernels
__global__ __launch_bounds__(256) void k_prep_wctxT(const float* __restrict__ W_ctx,
                                                    __bf16* __restrict__ WctxT) {
  int n = blockIdx.x * 256 + threadIdx.x;          // 0 .. 1024*1024-1
  int o = n >> 10, i = n & 1023;
  WctxT[(size_t)i * 1024 + o] = (__bf16)W_ctx[n];  // [k=i][n=o]
}

__global__ __launch_bounds__(256) void k_prep_amatT(const float* __restrict__ Wp1,
                                                    __bf16* __restrict__ AmatT) {
  int n = blockIdx.x * 256 + threadIdx.x;          // 0 .. 1024*256-1
  int o = n >> 8, p = n & 255;
  AmatT[n] = (p < 250) ? (__bf16)Wp1[(size_t)p * 3072 + o] : (__bf16)0.0f;
}

__global__ __launch_bounds__(256) void k_zero(float* __restrict__ p, int n) {
  int i = blockIdx.x * 256 + threadIdx.x;
  if (i < n) p[i] = 0.0f;
}

// ---------------------------------------------------------------- LSTM cell
// block = hidden unit j; 4 waves, wave q computes gate row j + q*1024 with float4 loads + shfl reduce
__global__ __launch_bounds__(128) void k_lstm(const float* __restrict__ dec_inp,
                                              const float* __restrict__ h_prev_h,
                                              const float* __restrict__ h_prev_c,
                                              const float* __restrict__ W_ih,
                                              const float* __restrict__ W_hh,
                                              const float* __restrict__ b_ih,
                                              const float* __restrict__ b_hh,
                                              float* __restrict__ out_h,
                                              float* __restrict__ out_c,
                                              float* __restrict__ hbuf) {
  __shared__ float gl[4];
  const int j = blockIdx.x;
  const int wave = threadIdx.x >> 5, lane = threadIdx.x & 31;
  const int row = j + wave * 1024;
  const float4* wi4 = (const float4*)(W_ih + (size_t)row * 2816);   // 704 float4
  const float4* wh4 = (const float4*)(W_hh + (size_t)row * 1024);   // 256 float4
  const float4* x4  = (const float4*)dec_inp;
  const float4* h4  = (const float4*)h_prev_h;
  float p = 0.f;
  for (int k = lane; k < 704; k += 32) p += dot4(x4[k], wi4[k]);
  for (int k = lane; k < 256; k += 32) p += dot4(h4[k], wh4[k]);
  p = wave_reduce(p);
  if (lane == 0) gl[wave] = p + b_ih[row] + b_hh[row];
  __syncthreads();
  if (threadIdx.x == 0) {
    float ig = 1.f / (1.f + expf(-gl[0]));
    float fg = 1.f / (1.f + expf(-gl[1]));
    float gg = tanhf(gl[2]);
    float og = 1.f / (1.f + expf(-gl[3]));
    float c  = fg * h_prev_c[j] + ig * gg;
    float h  = og * tanhf(c);
    out_h[j] = h; out_c[j] = c; hbuf[j] = h;
  }
}

// ---------------------------------------------------------------- dh / wq matvecs + tanh  (wave per row; 8 rows/block)
__global__ __launch_bounds__(256) void k_vec1(const float* __restrict__ cur_men,
                                              const float* __restrict__ hbuf,
                                              const float* __restrict__ W_info,
                                              const float* __restrict__ W_q,
                                              const float* __restrict__ b_q,
                                              float* __restrict__ tdh,
                                              float* __restrict__ twq) {
  const int wave = threadIdx.x >> 5, lane = threadIdx.x & 31;
  const int b = blockIdx.x * 8 + wave;           // 0 .. 2047
  const float4 *x4, *w4; int row;
  if (b < 1024) { x4 = (const float4*)cur_men; w4 = (const float4*)(W_info + (size_t)b * 1024); row = b; }
  else          { x4 = (const float4*)hbuf;    w4 = (const float4*)(W_q + (size_t)(b - 1024) * 1024); row = b - 1024; }
  float p = 0.f;
  for (int k = lane; k < 256; k += 32) p += dot4(x4[k], w4[k]);
  p = wave_reduce(p);
  if (lane == 0) {
    if (b < 1024) tdh[row] = tanhf(p);
    else          twq[row] = tanhf(p + b_q[row]);
  }
}

// v[p] = tanh(wq) . B[p] + tanh(dh) . C[p] + bp1[p]   (pad to 256; wave per row)
__global__ __launch_bounds__(256) void k_vec2(const float* __restrict__ twq,
                                              const float* __restrict__ tdh,
                                              const float* __restrict__ Wp1,
                                              const float* __restrict__ bp1,
                                              float* __restrict__ vbuf) {
  const int wave = threadIdx.x >> 5, lane = threadIdx.x & 31;
  const int p = blockIdx.x * 8 + wave;           // 0 .. 255
  float acc = 0.f;
  if (p < 250) {
    const float4* B4 = (const float4*)(Wp1 + (size_t)p * 3072 + 1024);
    const float4* C4 = (const float4*)(Wp1 + (size_t)p * 3072 + 2048);
    const float4* q4 = (const float4*)twq;
    const float4* d4 = (const float4*)tdh;
    for (int k = lane; k < 256; k += 32) acc += dot4(q4[k], B4[k]) + dot4(d4[k], C4[k]);
  }
  acc = wave_reduce(acc);
  if (lane == 0) vbuf[p] = (p < 250) ? acc + bp1[p] : 0.f;
}

// ---------------------------------------------------------------- fused WMMA: scores
// 32 s-rows per WG:  uh = enc @ WctxT ; t = tanh(uh) ; h1 = t @ AmatT + v ; score = relu(h1).Wp2 + bp2
__global__ __launch_bounds__(256, 1) void k_attn_scores(const float* __restrict__ enc_hs,
                                                        const __bf16* __restrict__ WctxT,   // [1024][1024]
                                                        const __bf16* __restrict__ AmatT,   // [1024][256]
                                                        const float* __restrict__ vbuf,     // [256]
                                                        const float* __restrict__ Wp2,      // [250]
                                                        const float* __restrict__ bp2,      // [1]
                                                        const int* __restrict__ src_mask,   // [S]
                                                        float* __restrict__ scores) {       // [S]
  __shared__ __align__(16) __bf16 encT[32 * 1024];   // 64 KB
  __shared__ __align__(16) __bf16 tB[32 * 256];      // 16 KB (quarter of o-range)
  __shared__ float scoreLDS[32];

  const int tid = threadIdx.x;
  const int wave = tid >> 5;
  const int lane = tid & 31;
  const int s0 = blockIdx.x * 32;

  for (int e = tid; e < 32 * 1024; e += 256) {
    int m = e >> 10, k = e & 1023;
    encT[e] = (__bf16)enc_hs[(size_t)(s0 + m) * 1024 + k];
  }
  if (tid < 32) scoreLDS[tid] = 0.f;
  __syncthreads();

  const int m    = lane & 15;          // A-frag row / D column select
  const int koff = (lane >> 4) * 8;    // A-frag K offset for lane half
  const int row0 = (lane >> 4) * 8;    // C/D: lanes 16-31 hold M+8
  const int p0a  = (wave * 2) * 16;    // this wave's h1 column tiles
  const int p0b  = (wave * 2 + 1) * 16;

  v8f h1_00 = {}, h1_01 = {}, h1_10 = {}, h1_11 = {};  // [msub][ptile]

  for (int quarter = 0; quarter < 4; ++quarter) {
    const int obase = quarter * 256;
#pragma unroll 1
    for (int t = 0; t < 2; ++t) {
      const int n0 = obase + (wave * 2 + t) * 16;
      v8f acc0 = {}, acc1 = {};
      for (int k0 = 0; k0 < 1024; k0 += 32) {
        const __bf16* bp = WctxT + (size_t)(k0 + koff) * 1024 + n0 + (lane & 15);
        bf16x16 b;
#pragma unroll
        for (int s = 0; s < 8; ++s) { b[s] = bp[(size_t)s * 1024]; b[s + 8] = bp[(size_t)(16 + s) * 1024]; }

        bf16x8 a0lo = *(const bf16x8*)(&encT[m * 1024 + k0 + koff]);
        bf16x8 a0hi = *(const bf16x8*)(&encT[m * 1024 + k0 + koff + 16]);
        bf16x16 a0 = CAT16(a0lo, a0hi);
        bf16x8 a1lo = *(const bf16x8*)(&encT[(m + 16) * 1024 + k0 + koff]);
        bf16x8 a1hi = *(const bf16x8*)(&encT[(m + 16) * 1024 + k0 + koff + 16]);
        bf16x16 a1 = CAT16(a1lo, a1hi);

        acc0 = __builtin_amdgcn_wmma_f32_16x16x32_bf16(false, a0, false, b, (short)0, acc0, false, false);
        acc1 = __builtin_amdgcn_wmma_f32_16x16x32_bf16(false, a1, false, b, (short)0, acc1, false, false);
      }
      const int col = (n0 - obase) + (lane & 15);
#pragma unroll
      for (int r = 0; r < 8; ++r) {
        tB[(row0 + r) * 256 + col]      = (__bf16)tanhf(acc0[r]);
        tB[(16 + row0 + r) * 256 + col] = (__bf16)tanhf(acc1[r]);
      }
    }
    __syncthreads();
    for (int k0 = 0; k0 < 256; k0 += 32) {
      bf16x8 a0lo = *(const bf16x8*)(&tB[m * 256 + k0 + koff]);
      bf16x8 a0hi = *(const bf16x8*)(&tB[m * 256 + k0 + koff + 16]);
      bf16x16 a0 = CAT16(a0lo, a0hi);
      bf16x8 a1lo = *(const bf16x8*)(&tB[(m + 16) * 256 + k0 + koff]);
      bf16x8 a1hi = *(const bf16x8*)(&tB[(m + 16) * 256 + k0 + koff + 16]);
      bf16x16 a1 = CAT16(a1lo, a1hi);

      const int okrow = obase + k0 + koff;
      const __bf16* bpa = AmatT + (size_t)okrow * 256 + p0a + (lane & 15);
      const __bf16* bpb = AmatT + (size_t)okrow * 256 + p0b + (lane & 15);
      bf16x16 b1, b2;
#pragma unroll
      for (int s = 0; s < 8; ++s) {
        b1[s] = bpa[s * 256]; b1[s + 8] = bpa[(16 + s) * 256];
        b2[s] = bpb[s * 256]; b2[s + 8] = bpb[(16 + s) * 256];
      }

      h1_00 = __builtin_amdgcn_wmma_f32_16x16x32_bf16(false, a0, false, b1, (short)0, h1_00, false, false);
      h1_01 = __builtin_amdgcn_wmma_f32_16x16x32_bf16(false, a0, false, b2, (short)0, h1_01, false, false);
      h1_10 = __builtin_amdgcn_wmma_f32_16x16x32_bf16(false, a1, false, b1, (short)0, h1_10, false, false);
      h1_11 = __builtin_amdgcn_wmma_f32_16x16x32_bf16(false, a1, false, b2, (short)0, h1_11, false, false);
    }
    __syncthreads();
  }

  {
    const int pA = p0a + (lane & 15);
    const int pB = p0b + (lane & 15);
    const float wA = (pA < 250) ? Wp2[pA] : 0.f;
    const float wB = (pB < 250) ? Wp2[pB] : 0.f;
    const float vA = vbuf[pA], vB = vbuf[pB];
#pragma unroll
    for (int r = 0; r < 8; ++r) {
      float c0 = fmaxf(h1_00[r] + vA, 0.f) * wA + fmaxf(h1_01[r] + vB, 0.f) * wB;
      float c1 = fmaxf(h1_10[r] + vA, 0.f) * wA + fmaxf(h1_11[r] + vB, 0.f) * wB;
      atomicAdd(&scoreLDS[row0 + r], c0);
      atomicAdd(&scoreLDS[16 + row0 + r], c1);
    }
  }
  __syncthreads();
  if (tid < 32) {
    int s = s0 + tid;
    float sc = scoreLDS[tid] + bp2[0];
    scores[s] = (src_mask[s] != 0) ? NEG_INF : sc;
  }
}

// ---------------------------------------------------------------- single-block softmax (inputs pre-masked with -inf)
__global__ __launch_bounds__(1024) void k_softmax(const float* __restrict__ in,
                                                  float* __restrict__ out, int n) {
  __shared__ float red[1024];
  int tid = threadIdx.x;
  float mx = NEG_INF;
  for (int i = tid; i < n; i += 1024) mx = fmaxf(mx, in[i]);
  red[tid] = mx; __syncthreads();
  for (int s = 512; s > 0; s >>= 1) { if (tid < s) red[tid] = fmaxf(red[tid], red[tid + s]); __syncthreads(); }
  mx = red[0]; __syncthreads();
  float sum = 0.f;
  for (int i = tid; i < n; i += 1024) sum += expf(in[i] - mx);
  red[tid] = sum; __syncthreads();
  for (int s = 512; s > 0; s >>= 1) { if (tid < s) red[tid] += red[tid + s]; __syncthreads(); }
  float inv = 1.f / red[0];
  for (int i = tid; i < n; i += 1024) out[i] = expf(in[i] - mx) * inv;
}

// ---------------------------------------------------------------- ctx = sum_s attn[s] * enc_hs[s,:]  (float4 columns)
__global__ __launch_bounds__(256) void k_ctx(const float* __restrict__ enc_hs,
                                             const float* __restrict__ attn,
                                             float* __restrict__ ctx) {
  int tid = threadIdx.x;
  float4 acc = {0.f, 0.f, 0.f, 0.f};
  int r0 = blockIdx.x * 256;
  for (int r = 0; r < 256; ++r) {
    int s = r0 + r;
    float a = attn[s];
    const float4* row4 = (const float4*)(enc_hs + (size_t)s * 1024);
    float4 v = row4[tid];
    acc.x += a * v.x; acc.y += a * v.y; acc.z += a * v.z; acc.w += a * v.w;
  }
  atomicAdd(&ctx[tid * 4 + 0], acc.x);
  atomicAdd(&ctx[tid * 4 + 1], acc.y);
  atomicAdd(&ctx[tid * 4 + 2], acc.z);
  atomicAdd(&ctx[tid * 4 + 3], acc.w);
}

// ---------------------------------------------------------------- logits GEMV (16384 x 3072): wave per row, b128 loads, shfl reduce
__global__ __launch_bounds__(256) void k_logits(const float* __restrict__ ctx,
                                                const float* __restrict__ hbuf,
                                                const float* __restrict__ cur_men,
                                                const float* __restrict__ W_out,
                                                const float* __restrict__ b_out,
                                                const int* __restrict__ src_mask,
                                                float* __restrict__ logits) {
  const int wave = threadIdx.x >> 5, lane = threadIdx.x & 31;
  const int s = blockIdx.x * 8 + wave;
  const float4* w4 = (const float4*)(W_out + (size_t)s * 3072);    // 768 float4
  const float4* c4 = (const float4*)ctx;
  const float4* h4 = (const float4*)hbuf;
  const float4* m4 = (const float4*)cur_men;
  float p = 0.f;
  for (int k = lane; k < 256; k += 32) p += dot4(c4[k], w4[k]);
  for (int k = lane; k < 256; k += 32) p += dot4(h4[k], w4[256 + k]);
  for (int k = lane; k < 256; k += 32) p += dot4(m4[k], w4[512 + k]);
  p = wave_reduce(p);
  if (lane == 0)
    logits[s] = (src_mask[s] != 0) ? NEG_INF : p + b_out[s];
}

// ---------------------------------------------------------------- launcher
extern "C" void kernel_launch(void* const* d_in, const int* in_sizes, int n_in,
                              void* d_out, int out_size, void* d_ws, size_t ws_size,
                              hipStream_t stream) {
  const float* dec_inp  = (const float*)d_in[0];
  const float* h_prev_h = (const float*)d_in[1];
  const float* h_prev_c = (const float*)d_in[2];
  const float* enc_hs   = (const float*)d_in[3];
  const float* cur_men  = (const float*)d_in[4];
  const int*   src_mask = (const int*)d_in[5];
  const float* W_ih     = (const float*)d_in[6];
  const float* W_hh     = (const float*)d_in[7];
  const float* b_ih     = (const float*)d_in[8];
  const float* b_hh     = (const float*)d_in[9];
  const float* W_info   = (const float*)d_in[10];
  const float* W_ctx    = (const float*)d_in[11];
  const float* W_q      = (const float*)d_in[12];
  const float* b_q      = (const float*)d_in[13];
  const float* Wp1      = (const float*)d_in[14];
  const float* bp1      = (const float*)d_in[15];
  const float* Wp2      = (const float*)d_in[16];
  const float* bp2      = (const float*)d_in[17];
  const float* W_out    = (const float*)d_in[18];
  const float* b_out    = (const float*)d_in[19];

  char* ws = (char*)d_ws;
  __bf16* WctxT  = (__bf16*)(ws + 0);          // 2 MB
  __bf16* AmatT  = (__bf16*)(ws + 2097152);    // 512 KB
  float* hbuf    = (float*)(ws + 2621440);     // 4 KB
  float* tdh     = (float*)(ws + 2625536);     // 4 KB
  float* twq     = (float*)(ws + 2629632);     // 4 KB
  float* vbuf    = (float*)(ws + 2633728);     // 1 KB
  float* scores  = (float*)(ws + 2634752);     // 64 KB
  float* attn    = (float*)(ws + 2700288);     // 64 KB
  float* ctxb    = (float*)(ws + 2765824);     // 4 KB
  float* logits  = (float*)(ws + 2769920);     // 64 KB

  float* out_h   = (float*)d_out;          // [1024]
  float* out_c   = out_h + 1024;           // [1024]
  float* att_out = out_h + 2048;           // [16384]

  k_prep_wctxT<<<4096, 256, 0, stream>>>(W_ctx, WctxT);
  k_prep_amatT<<<1024, 256, 0, stream>>>(Wp1, AmatT);
  k_zero<<<4, 256, 0, stream>>>(ctxb, 1024);
  k_lstm<<<1024, 128, 0, stream>>>(dec_inp, h_prev_h, h_prev_c, W_ih, W_hh, b_ih, b_hh,
                                   out_h, out_c, hbuf);
  k_vec1<<<256, 256, 0, stream>>>(cur_men, hbuf, W_info, W_q, b_q, tdh, twq);
  k_vec2<<<32, 256, 0, stream>>>(twq, tdh, Wp1, bp1, vbuf);
  k_attn_scores<<<512, 256, 0, stream>>>(enc_hs, WctxT, AmatT, vbuf, Wp2, bp2, src_mask, scores);
  k_softmax<<<1, 1024, 0, stream>>>(scores, attn, 16384);
  k_ctx<<<64, 256, 0, stream>>>(enc_hs, attn, ctxb);
  k_logits<<<2048, 256, 0, stream>>>(ctxb, hbuf, cur_men, W_out, b_out, src_mask, logits);
  k_softmax<<<1, 1024, 0, stream>>>(logits, att_out, 16384);
}
